// GraphAttentionLayer_29901562315387
// MI455X (gfx1250) — matrix operand order repaired and checked by reading, not compiled, not run on hardware
//
#include <hip/hip_runtime.h>
#include <hip/hip_bf16.h>

#define N_NODES 50000
#define N_EDGES 640000
#define IN_DIM 128
#define HEADS 4
#define HEAD_DIM 32
#define NEG_SLOPE 0.2f
#define LN_EPS 1e-5f
#define M_TILES (N_NODES / 16)   // 3125 exactly

typedef __attribute__((ext_vector_type(16))) __bf16 v16bf;
typedef __attribute__((ext_vector_type(8)))  float  v8f;

static __device__ __forceinline__ __bf16 f2bf(float f) {
  unsigned u = __float_as_uint(f);
  u += 0x7FFFu + ((u >> 16) & 1u);           // round-to-nearest-even
  unsigned short s = (unsigned short)(u >> 16);
  return __builtin_bit_cast(__bf16, s);
}

// monotonic float <-> uint map so atomicMax(u32) implements float max
static __device__ __forceinline__ unsigned fenc(float f) {
  unsigned u = __float_as_uint(f);
  return (u & 0x80000000u) ? ~u : (u | 0x80000000u);
}
static __device__ __forceinline__ float fdec(unsigned u) {
  return __uint_as_float((u & 0x80000000u) ? (u & 0x7FFFFFFFu) : ~u);
}

// ---------------------------------------------------------------------------
// 0) init: agg (= d_out) -> 0, denom -> 0, emax -> 0 (below enc(-inf))
__global__ void k_init(float* __restrict__ agg, float* __restrict__ denom,
                       unsigned* __restrict__ emax) {
  int i = blockIdx.x * blockDim.x + threadIdx.x;
  if (i < N_NODES * IN_DIM) agg[i] = 0.0f;
  if (i < N_NODES * HEADS) { denom[i] = 0.0f; emax[i] = 0u; }
}

// ---------------------------------------------------------------------------
// 1) h = x @ W  via v_wmma_f32_16x16x32_bf16.
//    Block = 8 waves; each wave owns one 16-row M tile x all 128 N columns
//    (8 accumulator tiles). W is staged once per block into LDS as bf16 in
//    fragment-major order: wl[(t*4+kk)*32*16 + lane*16 + j] = W[k][n] with
//    k = kk*32 + (j>=8 ? 16 : 0) + (lane>=16 ? 8 : 0) + (j&7), n = t*16 + (lane&15)  (B layout).
__global__ __launch_bounds__(256)
void k_gemm(const float* __restrict__ x, const float* __restrict__ W,
            float* __restrict__ h) {
  __shared__ __bf16 wl[8 * 4 * 32 * 16];   // 16384 bf16 = 32 KB

  const int tid = threadIdx.x;
  for (int idx = tid; idx < 8 * 4 * 32 * 16; idx += 256) {
    int j    = idx & 15;
    int lane = (idx >> 4) & 31;
    int kk   = (idx >> 9) & 3;
    int t    = idx >> 11;
    int k = kk * 32 + ((j >> 3) << 4) + ((lane >> 4) << 3) + (j & 7);
    int n = t * 16 + (lane & 15);
    wl[idx] = f2bf(W[k * IN_DIM + n]);
  }
  __syncthreads();

  const int wave = tid >> 5;
  const int lane = tid & 31;
  const int mtile = blockIdx.x * 8 + wave;
  if (mtile >= M_TILES) return;            // wave-uniform: EXEC stays all-ones

  const int row  = mtile * 16 + (lane & 15);
  const int koff = (lane < 16) ? 0 : 8;    // A-matrix 16x32 bf16 lane layout

  v8f acc[8] = {};

#pragma unroll
  for (int kk = 0; kk < 4; kk++) {
    const float* xr = x + (size_t)row * IN_DIM + kk * 32 + koff;
    float4 p0 = *(const float4*)(xr + 0);
    float4 p1 = *(const float4*)(xr + 4);
    float4 p2 = *(const float4*)(xr + 16);
    float4 p3 = *(const float4*)(xr + 20);
    v16bf a;
    a[0]  = f2bf(p0.x); a[1]  = f2bf(p0.y); a[2]  = f2bf(p0.z); a[3]  = f2bf(p0.w);
    a[4]  = f2bf(p1.x); a[5]  = f2bf(p1.y); a[6]  = f2bf(p1.z); a[7]  = f2bf(p1.w);
    a[8]  = f2bf(p2.x); a[9]  = f2bf(p2.y); a[10] = f2bf(p2.z); a[11] = f2bf(p2.w);
    a[12] = f2bf(p3.x); a[13] = f2bf(p3.y); a[14] = f2bf(p3.z); a[15] = f2bf(p3.w);

#pragma unroll
    for (int t = 0; t < 8; t++) {
      v16bf b = *(const v16bf*)&wl[(((t * 4 + kk) * 32) + lane) * 16];
      acc[t] = __builtin_amdgcn_wmma_f32_16x16x32_bf16(
          false, a, false, b, (short)0, acc[t], false, false);
    }
  }

  // C/D layout: VGPR j -> M = j (+8 for lanes 16..31), N = t*16 + (lane&15)
  const int rbase = mtile * 16 + ((lane >> 4) << 3);
  const int col   = lane & 15;
#pragma unroll
  for (int j = 0; j < 8; j++) {
    float* hr = h + (size_t)(rbase + j) * IN_DIM + col;
#pragma unroll
    for (int t = 0; t < 8; t++) hr[t * 16] = acc[t][j];
  }
}

// ---------------------------------------------------------------------------
// 2) a_src/a_dst: wave per node, 8 lanes per head, shuffle-reduce over head
__global__ __launch_bounds__(256)
void k_att(const float* __restrict__ h, const float* __restrict__ att_s,
           const float* __restrict__ att_d, float* __restrict__ a_src,
           float* __restrict__ a_dst) {
  int wave = threadIdx.x >> 5, lane = threadIdx.x & 31;
  int node = blockIdx.x * 8 + wave;
  if (node >= N_NODES) return;
  int head = lane >> 3;
  int dsub = (lane & 7) * 4;
  float4 hv = *(const float4*)(h + (size_t)node * IN_DIM + lane * 4);
  float4 as = *(const float4*)(att_s + head * HEAD_DIM + dsub);
  float4 ad = *(const float4*)(att_d + head * HEAD_DIM + dsub);
  float ps = hv.x * as.x + hv.y * as.y + hv.z * as.z + hv.w * as.w;
  float pd = hv.x * ad.x + hv.y * ad.y + hv.z * ad.z + hv.w * ad.w;
#pragma unroll
  for (int m = 1; m < 8; m <<= 1) {
    ps += __shfl_xor(ps, m, 32);
    pd += __shfl_xor(pd, m, 32);
  }
  if ((lane & 7) == 0) {
    a_src[node * HEADS + head] = ps;
    a_dst[node * HEADS + head] = pd;
  }
}

// ---------------------------------------------------------------------------
// 3) segment max of leaky_relu(a_src[s]+a_dst[d]) over dst (incl self-loops)
__global__ void k_emax(const int* __restrict__ ei, const float* __restrict__ a_src,
                       const float* __restrict__ a_dst, unsigned* __restrict__ emax) {
  int i = blockIdx.x * blockDim.x + threadIdx.x;
  if (i >= N_EDGES + N_NODES) return;
  int s, d;
  if (i < N_EDGES) { s = ei[i]; d = ei[N_EDGES + i]; } else { s = d = i - N_EDGES; }
#pragma unroll
  for (int hh = 0; hh < HEADS; hh++) {
    float e = a_src[s * HEADS + hh] + a_dst[d * HEADS + hh];
    e = (e >= 0.0f) ? e : NEG_SLOPE * e;
    atomicMax(&emax[d * HEADS + hh], fenc(e));
  }
}

// 4) denom = segment_sum exp(e - emax[d])
__global__ void k_esum(const int* __restrict__ ei, const float* __restrict__ a_src,
                       const float* __restrict__ a_dst, const unsigned* __restrict__ emax,
                       float* __restrict__ denom) {
  int i = blockIdx.x * blockDim.x + threadIdx.x;
  if (i >= N_EDGES + N_NODES) return;
  int s, d;
  if (i < N_EDGES) { s = ei[i]; d = ei[N_EDGES + i]; } else { s = d = i - N_EDGES; }
#pragma unroll
  for (int hh = 0; hh < HEADS; hh++) {
    float e = a_src[s * HEADS + hh] + a_dst[d * HEADS + hh];
    e = (e >= 0.0f) ? e : NEG_SLOPE * e;
    atomicAdd(&denom[d * HEADS + hh], expf(e - fdec(emax[d * HEADS + hh])));
  }
}

// 5) weighted scatter: agg[d] += alpha * h[s]. Wave per edge, float4 per lane;
//    agg (= d_out, 25.6 MB) and h live in the 192 MB L2, so gathers + f32
//    atomics stay on-chip.
__global__ __launch_bounds__(256)
void k_scatter(const int* __restrict__ ei, const float* __restrict__ h,
               const float* __restrict__ a_src, const float* __restrict__ a_dst,
               const unsigned* __restrict__ emax, const float* __restrict__ denom,
               float* __restrict__ agg) {
  int wave = threadIdx.x >> 5, lane = threadIdx.x & 31;
  long i = (long)blockIdx.x * 8 + wave;
  if (i >= (long)(N_EDGES + N_NODES)) return;
  int s, d;
  if (i < N_EDGES) { s = ei[i]; d = ei[N_EDGES + i]; } else { s = d = (int)(i - N_EDGES); }
  int hh = lane >> 3;
  float e = a_src[s * HEADS + hh] + a_dst[d * HEADS + hh];
  e = (e >= 0.0f) ? e : NEG_SLOPE * e;
  float alpha = expf(e - fdec(emax[d * HEADS + hh])) / (denom[d * HEADS + hh] + 1e-16f);
  float4 hv = *(const float4*)(h + (size_t)s * IN_DIM + lane * 4);
  float* o = agg + (size_t)d * IN_DIM + lane * 4;
  atomicAdd(o + 0, hv.x * alpha);
  atomicAdd(o + 1, hv.y * alpha);
  atomicAdd(o + 2, hv.z * alpha);
  atomicAdd(o + 3, hv.w * alpha);
}

// ---------------------------------------------------------------------------
// 6) out = LN(x + agg + bias) * gamma + beta  (wave per row, wave32 reduce)
__global__ __launch_bounds__(256)
void k_final(const float* __restrict__ x, const float* __restrict__ bias,
             const float* __restrict__ gamma, const float* __restrict__ beta,
             float* __restrict__ out) {
  int wave = threadIdx.x >> 5, lane = threadIdx.x & 31;
  int node = blockIdx.x * 8 + wave;
  if (node >= N_NODES) return;
  float4 o  = *(const float4*)(out + (size_t)node * IN_DIM + lane * 4);
  float4 xv = *(const float4*)(x + (size_t)node * IN_DIM + lane * 4);
  float4 bv = *(const float4*)(bias + lane * 4);
  float v0 = o.x + xv.x + bv.x, v1 = o.y + xv.y + bv.y;
  float v2 = o.z + xv.z + bv.z, v3 = o.w + xv.w + bv.w;
  float sum = v0 + v1 + v2 + v3;
  float sq  = v0 * v0 + v1 * v1 + v2 * v2 + v3 * v3;
#pragma unroll
  for (int m = 1; m < 32; m <<= 1) {
    sum += __shfl_xor(sum, m, 32);
    sq  += __shfl_xor(sq, m, 32);
  }
  float mean = sum * (1.0f / IN_DIM);
  float var  = sq * (1.0f / IN_DIM) - mean * mean;
  float rstd = rsqrtf(var + LN_EPS);
  float4 gv = *(const float4*)(gamma + lane * 4);
  float4 be = *(const float4*)(beta + lane * 4);
  float4 r;
  r.x = (v0 - mean) * rstd * gv.x + be.x;
  r.y = (v1 - mean) * rstd * gv.y + be.y;
  r.z = (v2 - mean) * rstd * gv.z + be.z;
  r.w = (v3 - mean) * rstd * gv.w + be.w;
  *(float4*)(out + (size_t)node * IN_DIM + lane * 4) = r;
}

// ---------------------------------------------------------------------------
extern "C" void kernel_launch(void* const* d_in, const int* in_sizes, int n_in,
                              void* d_out, int out_size, void* d_ws, size_t ws_size,
                              hipStream_t stream) {
  const float* x       = (const float*)d_in[0];
  const int*   ei      = (const int*)d_in[1];   // [2, E] int32 (JAX x64 off)
  const float* W       = (const float*)d_in[2];
  const float* att_src = (const float*)d_in[3];
  const float* att_dst = (const float*)d_in[4];
  const float* bias    = (const float*)d_in[5];
  const float* gamma   = (const float*)d_in[6];
  const float* beta    = (const float*)d_in[7];
  float* out = (float*)d_out;                   // agg accumulator, then result

  float*    ws    = (float*)d_ws;
  float*    h     = ws;                                        // N*128 f32
  float*    a_src = ws + (size_t)N_NODES * IN_DIM;             // N*4
  float*    a_dst = a_src + (size_t)N_NODES * HEADS;           // N*4
  unsigned* emax  = (unsigned*)(a_dst + (size_t)N_NODES * HEADS); // N*4
  float*    denom = (float*)(emax + (size_t)N_NODES * HEADS);  // N*4

  const int EE = N_EDGES + N_NODES;

  k_init<<<(N_NODES * IN_DIM + 255) / 256, 256, 0, stream>>>(out, denom, emax);
  k_gemm<<<(M_TILES + 7) / 8, 256, 0, stream>>>(x, W, h);
  k_att<<<(N_NODES + 7) / 8, 256, 0, stream>>>(h, att_src, att_dst, a_src, a_dst);
  k_emax<<<(EE + 255) / 256, 256, 0, stream>>>(ei, a_src, a_dst, emax);
  k_esum<<<(EE + 255) / 256, 256, 0, stream>>>(ei, a_src, a_dst, emax, denom);
  k_scatter<<<(EE + 7) / 8, 256, 0, stream>>>(ei, h, a_src, a_dst, emax, denom, out);
  k_final<<<(N_NODES + 7) / 8, 256, 0, stream>>>(x, bias, gamma, beta, out);
}
